// MultiHeadScalarAttention_33629593927874
// MI455X (gfx1250) — matrix operand order repaired and checked by reading, not compiled, not run on hardware
//
#include <hip/hip_runtime.h>

typedef __bf16 bf16_t;
typedef __attribute__((ext_vector_type(8)))  __bf16 v8bf;
typedef __attribute__((ext_vector_type(16))) __bf16 v16bf;
typedef __attribute__((ext_vector_type(8)))  float  v8f;
typedef __attribute__((ext_vector_type(4)))  int    v4i_t;

#define AS1 __attribute__((address_space(1)))
#define AS3 __attribute__((address_space(3)))

// Async global->LDS copy (CDNA5 GLOBAL_LOAD_ASYNC_TO_LDS_B128, ASYNCcnt).
#if defined(__has_builtin)
#if __has_builtin(__builtin_amdgcn_global_load_async_to_lds_b128) && \
    __has_builtin(__builtin_amdgcn_s_wait_asynccnt)
#define USE_ASYNC_LDS 1
#endif
#if __has_builtin(__builtin_amdgcn_update_dpp)
#define USE_DPP_RED 1
#endif
#endif

// ---------------------------------------------------------------------------
// Cross-lane reductions over 16-lane halves (wave32).
// xor basis {1,2,7,15}: after each step values are uniform per group, so
// quad_perm(xor1), quad_perm(xor2), row_half_mirror(xor7), row_mirror(xor15)
// reduce a full 16-lane group using DPP only (no LDS bpermute).
// ---------------------------------------------------------------------------
#if defined(USE_DPP_RED)
template <int CTRL>
__device__ __forceinline__ float dpp_xor_f(float x) {
  int v = __builtin_amdgcn_update_dpp(0, __float_as_int(x), CTRL, 0xf, 0xf, true);
  return __int_as_float(v);
}
__device__ __forceinline__ float red_max16(float x) {
  x = fmaxf(x, dpp_xor_f<0x0B1>(x));   // quad_perm [1,0,3,2]  (xor 1)
  x = fmaxf(x, dpp_xor_f<0x04E>(x));   // quad_perm [2,3,0,1]  (xor 2)
  x = fmaxf(x, dpp_xor_f<0x141>(x));   // row_half_mirror      (xor 7)
  x = fmaxf(x, dpp_xor_f<0x140>(x));   // row_mirror           (xor 15)
  return x;
}
__device__ __forceinline__ float red_sum16(float x) {
  x += dpp_xor_f<0x0B1>(x);
  x += dpp_xor_f<0x04E>(x);
  x += dpp_xor_f<0x141>(x);
  x += dpp_xor_f<0x140>(x);
  return x;
}
#else
__device__ __forceinline__ float red_max16(float x) {
  x = fmaxf(x, __shfl_xor(x, 1));
  x = fmaxf(x, __shfl_xor(x, 2));
  x = fmaxf(x, __shfl_xor(x, 4));
  x = fmaxf(x, __shfl_xor(x, 8));
  return x;
}
__device__ __forceinline__ float red_sum16(float x) {
  x += __shfl_xor(x, 1);
  x += __shfl_xor(x, 2);
  x += __shfl_xor(x, 4);
  x += __shfl_xor(x, 8);
  return x;
}
#endif

// ---------------------------------------------------------------------------
// WMMA helpers (CDNA5 wave32 layouts, ISA 7.12.2)
// ---------------------------------------------------------------------------
__device__ __forceinline__ v16bf cat16(v8bf lo, v8bf hi) {
  return __builtin_shufflevector(lo, hi, 0, 1, 2, 3, 4, 5, 6, 7,
                                         8, 9, 10, 11, 12, 13, 14, 15);
}

// A-fragment (16x32 bf16) from a bf16 row.  Element i holds
// K = i + (i>=8 ? 8:0) + koff, koff = (lane>=16 ? 8 : 0).
__device__ __forceinline__ v16bf load_frag_a(const bf16_t* row, int koff) {
  v8bf lo = *(const v8bf*)(row + koff);
  v8bf hi = *(const v8bf*)(row + koff + 16);
  return cat16(lo, hi);
}

// A-fragment built from an fp32 row (LDS staged raw by the async copy);
// converts to bf16 at fragment-build time.
__device__ __forceinline__ v16bf frag_a_from_f32(const float* row, int koff) {
  v8f lo = *(const v8f*)(row + koff);
  v8f hi = *(const v8f*)(row + koff + 16);
  v16bf r;
#pragma unroll
  for (int i = 0; i < 8; ++i) r[i] = (bf16_t)lo[i];
#pragma unroll
  for (int i = 0; i < 8; ++i) r[8 + i] = (bf16_t)hi[i];
  return r;
}

// B-fragment (32x16 bf16): lane's column (N = lane%16); element i holds
// K = i + (i>=8 ? 8:0) + boff/2, boff = (lane>=16 ? 16 : 0).
__device__ __forceinline__ v16bf load_frag_b(const bf16_t* col, int boff) {
  v8bf lo = *(const v8bf*)(col + boff / 2);
  v8bf hi = *(const v8bf*)(col + boff / 2 + 16);
  return cat16(lo, hi);
}

__device__ __forceinline__ v8f wmma_bf16(v16bf a, v16bf b, v8f c) {
  return __builtin_amdgcn_wmma_f32_16x16x32_bf16(
      false, a, false, b, (short)0, c, false, false);
}

__device__ __forceinline__ v8f v8f_zero() {
  v8f z;
#pragma unroll
  for (int i = 0; i < 8; ++i) z[i] = 0.0f;
  return z;
}

// ---------------------------------------------------------------------------
// Problem constants
// ---------------------------------------------------------------------------
#define BATCH 4
#define SEQ   2048
#define DIM   1024
#define HEADS 16
#define HDIM  64

// ---------------------------------------------------------------------------
// Kernel 1: Y = relu(X @ W + b) [* 1/8 for Q], fp32 in, bf16 out.
// Block tile 128x128, BK=32; 128 threads = 4 waves (2x2), wave tile 64x64
// (16 WMMAs per wave per k-step).  Double-buffered LDS: async-DMA of X(k+1)
// and global loads of W(k+1) overlap the WMMAs of tile k.  One barrier/iter.
// ---------------------------------------------------------------------------
#define BM 128
#define BN 128
#define BK 32
#define NKSTEP (DIM / BK)
#define XFSTR 36   // fp32 elems per LDS row (144B: 16B-aligned, conflict-free)
#define WSTR  40   // bf16 elems per LDS row (80B)

// Issue X-tile [128 x 32] fp32 copy into LDS buffer (async when available).
__device__ __forceinline__ void stage_x_tile(const float* __restrict__ X,
                                             int m0, int k0, float* Xbuf,
                                             int tid) {
#pragma unroll
  for (int i = 0; i < 8; ++i) {
    int id = tid + i * 128;
    int row = id >> 3;
    int c4 = (id & 7) * 4;
    const float* gp = X + (size_t)(m0 + row) * DIM + k0 + c4;
    float* lp = Xbuf + row * XFSTR + c4;
#if defined(USE_ASYNC_LDS)
    __builtin_amdgcn_global_load_async_to_lds_b128(
        (AS1 v4i_t*)gp, (AS3 v4i_t*)lp, 0, 0);
#else
    *(float4*)lp = *(const float4*)gp;
#endif
  }
}

// Load W-tile [32 x 128] fp32 into registers (global loads, waited lazily).
__device__ __forceinline__ void load_w_regs(const float* __restrict__ W,
                                            int n0, int k0, float4* wr,
                                            int tid) {
#pragma unroll
  for (int i = 0; i < 8; ++i) {
    int id = tid + i * 128;
    int kr = id >> 5;
    int c4 = (id & 31) * 4;
    wr[i] = *(const float4*)(W + (size_t)(k0 + kr) * DIM + n0 + c4);
  }
}

// Convert + transpose W registers into LDS buffer Ws[n][k].
__device__ __forceinline__ void store_w_tile(const float4* wr, bf16_t* Wbuf,
                                             int tid) {
#pragma unroll
  for (int i = 0; i < 8; ++i) {
    int id = tid + i * 128;
    int kr = id >> 5;
    int c4 = (id & 31) * 4;
    Wbuf[(c4 + 0) * WSTR + kr] = (bf16_t)wr[i].x;
    Wbuf[(c4 + 1) * WSTR + kr] = (bf16_t)wr[i].y;
    Wbuf[(c4 + 2) * WSTR + kr] = (bf16_t)wr[i].z;
    Wbuf[(c4 + 3) * WSTR + kr] = (bf16_t)wr[i].w;
  }
}

template <int TRANSPOSE, int SCALE_Q>
__global__ __launch_bounds__(128)
void qkv_gemm_kernel(const float* __restrict__ X, const float* __restrict__ W,
                     const float* __restrict__ bias, bf16_t* __restrict__ out) {
  __shared__ float  Xf[2][BM * XFSTR];   // raw fp32 A tiles (double buffer)
  __shared__ bf16_t Ws[2][BN * WSTR];    // transposed bf16 B tiles

  const int m0 = blockIdx.x * BM;
  const int n0 = blockIdx.y * BN;
  const int tid = threadIdx.x;
  const int lane = tid & 31;
  const int wv = tid >> 5;
  const int wm = wv >> 1;            // 0..1 -> 64-row stripe
  const int wn = wv & 1;             // 0..1 -> 64-col stripe
  const int lr = lane & 15;
  const bool hih = lane >= 16;
  const int koff = hih ? 8 : 0;
  const int boff = hih ? 16 : 0;

  v8f acc[4][4];
#pragma unroll
  for (int mi = 0; mi < 4; ++mi)
#pragma unroll
    for (int ni = 0; ni < 4; ++ni) acc[mi][ni] = v8f_zero();

  // --- prologue: stage tile 0 into buffer 0 ---
  float4 wr[8];
  stage_x_tile(X, m0, 0, Xf[0], tid);
  load_w_regs(W, n0, 0, wr, tid);
  store_w_tile(wr, Ws[0], tid);
#if defined(USE_ASYNC_LDS)
  __builtin_amdgcn_s_wait_asynccnt(0);
#endif
  __syncthreads();

  for (int kk = 0; kk < NKSTEP; ++kk) {
    const int cur = kk & 1;
    const int nxt = cur ^ 1;
    const bool has_next = (kk + 1) < NKSTEP;

    // Issue next tile's DMA + global loads before computing (overlap).
    if (has_next) {
      stage_x_tile(X, m0, (kk + 1) * BK, Xf[nxt], tid);
      load_w_regs(W, n0, (kk + 1) * BK, wr, tid);
    }

    // Compute tile kk from buffer `cur`.
    v16bf a[4], bfr[4];
#pragma unroll
    for (int mi = 0; mi < 4; ++mi)
      a[mi] = frag_a_from_f32(&Xf[cur][(wm * 64 + mi * 16 + lr) * XFSTR], koff);
#pragma unroll
    for (int ni = 0; ni < 4; ++ni)
      bfr[ni] = load_frag_b(&Ws[cur][(wn * 64 + ni * 16 + lr) * WSTR], boff);
    // A-major order: same A operand across 4 consecutive WMMAs (operand reuse)
#pragma unroll
    for (int mi = 0; mi < 4; ++mi)
#pragma unroll
      for (int ni = 0; ni < 4; ++ni)
        acc[mi][ni] = wmma_bf16(a[mi], bfr[ni], acc[mi][ni]);

    // Finish staging the next W tile, then sync once.
    if (has_next) store_w_tile(wr, Ws[nxt], tid);
#if defined(USE_ASYNC_LDS)
    __builtin_amdgcn_s_wait_asynccnt(0);
#endif
    __syncthreads();
  }

  // --- epilogue: bias + relu (+1/sqrt(HD) for Q) + bf16 store ---
#pragma unroll
  for (int mi = 0; mi < 4; ++mi) {
#pragma unroll
    for (int ni = 0; ni < 4; ++ni) {
      int n = n0 + wn * 64 + ni * 16 + lr;
      float bvv = bias[n];
      int h = n >> 6;          // n / HDIM
      int d = n & 63;          // n % HDIM
#pragma unroll
      for (int r = 0; r < 8; ++r) {
        int m = m0 + wm * 64 + mi * 16 + r + (hih ? 8 : 0);
        int bb = m >> 11;      // m / SEQ
        int ss = m & 2047;     // m % SEQ
        float v = acc[mi][ni][r] + bvv;
        v = v > 0.0f ? v : 0.0f;
        if (SCALE_Q) v *= 0.125f;   // fold softmax 1/sqrt(64) into Q
        size_t idx;
        if (TRANSPOSE)
          idx = ((size_t)((bb * HEADS + h) * HDIM + d)) * SEQ + ss;
        else
          idx = ((size_t)((bb * HEADS + h) * SEQ + ss)) * HDIM + d;
        out[idx] = (bf16_t)v;
      }
    }
  }
}

// ---------------------------------------------------------------------------
// Kernel 2: flash attention over bf16 Q[bh][s][d] (pre-scaled), K[bh][s][d],
// V[bh][d][s].  grid = (SEQ/64, B*H); block = 128 (4 waves); each wave owns
// 16 q-rows.  All fragments of a tile are loaded before their WMMAs; V loads
// are issued before the softmax VALU; reductions use DPP16 (no LDS permute).
// ---------------------------------------------------------------------------
#define PSTR 72   // 144B rows: conflict-free 16B LDS accesses

__global__ __launch_bounds__(128)
void flash_attn_kernel(const bf16_t* __restrict__ q,
                       const bf16_t* __restrict__ k,
                       const bf16_t* __restrict__ v,
                       float* __restrict__ out) {
  __shared__ bf16_t Ps[4][16][PSTR];

  const int lane = threadIdx.x & 31;
  const int wv = threadIdx.x >> 5;   // 0..3
  const int lr = lane & 15;
  const bool hih = lane >= 16;
  const int koff = hih ? 8 : 0;
  const int boff = hih ? 16 : 0;

  const int qt = blockIdx.x;         // 0..SEQ/64-1
  const int bh = blockIdx.y;         // 0..B*H-1
  const int qbase = qt * 64 + wv * 16;

  // Q A-fragments: register-resident for the whole KV loop.
  const bf16_t* qrow = q + ((size_t)bh * SEQ + qbase + lr) * HDIM;
  v16bf qa[2];
  qa[0] = load_frag_a(qrow + 0 * 32, koff);
  qa[1] = load_frag_a(qrow + 1 * 32, koff);

  const bf16_t* kmat = k + (size_t)bh * SEQ * HDIM;   // [s][d]
  const bf16_t* vmat = v + (size_t)bh * HDIM * SEQ;   // [d][s]

  float mrow[8], lsum[8];
#pragma unroll
  for (int r = 0; r < 8; ++r) { mrow[r] = -1e30f; lsum[r] = 0.0f; }

  v8f acco[4];
#pragma unroll
  for (int t = 0; t < 4; ++t) acco[t] = v8f_zero();

  for (int kv = 0; kv < SEQ; kv += 64) {
    // Prefetch next K/V tile (128B per row; 2 rows per lane covers 64 rows).
    if (kv + 64 < SEQ) {
      const bf16_t* kn = kmat + (size_t)(kv + 64) * HDIM;
      __builtin_prefetch(kn + (size_t)(lane * 2 + 0) * HDIM, 0, 1);
      __builtin_prefetch(kn + (size_t)(lane * 2 + 1) * HDIM, 0, 1);
      __builtin_prefetch(vmat + (size_t)(lane * 2 + 0) * SEQ + kv + 64, 0, 1);
      __builtin_prefetch(vmat + (size_t)(lane * 2 + 1) * SEQ + kv + 64, 0, 1);
    }

    // ---- load ALL K fragments, then 8 WMMAs (A held constant per group) ----
    v16bf kf[4][2];
#pragma unroll
    for (int t = 0; t < 4; ++t) {
      const bf16_t* krow = kmat + (size_t)(kv + t * 16 + lr) * HDIM;
      kf[t][0] = load_frag_b(krow, boff);
      kf[t][1] = load_frag_b(krow + 32, boff);
    }
    v8f sc[4];
#pragma unroll
    for (int t = 0; t < 4; ++t) sc[t] = v8f_zero();
#pragma unroll
    for (int f = 0; f < 2; ++f)
#pragma unroll
      for (int t = 0; t < 4; ++t)
        sc[t] = wmma_bf16(qa[f], kf[t][f], sc[t]);

    // ---- issue V fragment loads now; softmax VALU hides their latency ----
    v16bf vf[4][2];
#pragma unroll
    for (int td = 0; td < 4; ++td) {
      const bf16_t* vcol = vmat + (size_t)(td * 16 + lr) * SEQ + kv;
      vf[td][0] = load_frag_b(vcol, boff);
      vf[td][1] = load_frag_b(vcol + 32, boff);
    }

    // ---- online softmax (Q pre-scaled; row stats within 16-lane halves) ----
    float alpha[8];
#pragma unroll
    for (int r = 0; r < 8; ++r) {
      float tm = fmaxf(fmaxf(sc[0][r], sc[1][r]), fmaxf(sc[2][r], sc[3][r]));
      tm = red_max16(tm);
      float mnew = fmaxf(mrow[r], tm);
      alpha[r] = __expf(mrow[r] - mnew);
      mrow[r] = mnew;

      float rs = 0.0f;
#pragma unroll
      for (int t = 0; t < 4; ++t) {
        float p = __expf(sc[t][r] - mnew);
        sc[t][r] = p;
        rs += p;
        Ps[wv][r + (hih ? 8 : 0)][t * 16 + lr] = (bf16_t)p;
      }
      rs = red_sum16(rs);
      lsum[r] = lsum[r] * alpha[r] + rs;
    }
#pragma unroll
    for (int t = 0; t < 4; ++t)
#pragma unroll
      for (int r = 0; r < 8; ++r) acco[t][r] *= alpha[r];

    // ---- O += P @ V  (P re-read from per-wave LDS as A-fragments) ----
    const bf16_t* prow = &Ps[wv][lr][0];
    v16bf pa[2];
    pa[0] = load_frag_a(prow + 0 * 32, koff);
    pa[1] = load_frag_a(prow + 1 * 32, koff);

#pragma unroll
    for (int f = 0; f < 2; ++f)
#pragma unroll
      for (int td = 0; td < 4; ++td)
        acco[td] = wmma_bf16(pa[f], vf[td][f], acco[td]);
  }

  // ---- epilogue: O /= l, write fp32 [b, s, h*HD + d] ----
  const int bb = bh >> 4;
  const int hh = bh & 15;
  float inv[8];
#pragma unroll
  for (int r = 0; r < 8; ++r) inv[r] = 1.0f / lsum[r];
#pragma unroll
  for (int td = 0; td < 4; ++td) {
    int d = td * 16 + lr;
#pragma unroll
    for (int r = 0; r < 8; ++r) {
      int srow = qbase + r + (hih ? 8 : 0);
      out[((size_t)(bb * SEQ + srow)) * DIM + hh * HDIM + d] = acco[td][r] * inv[r];
    }
  }
}

// ---------------------------------------------------------------------------
// Launch
// ---------------------------------------------------------------------------
extern "C" void kernel_launch(void* const* d_in, const int* in_sizes, int n_in,
                              void* d_out, int out_size, void* d_ws, size_t ws_size,
                              hipStream_t stream) {
  const float* query = (const float*)d_in[0];
  const float* key_  = (const float*)d_in[1];
  const float* value = (const float*)d_in[2];
  const float* Wq = (const float*)d_in[3];
  const float* bq = (const float*)d_in[4];
  const float* Wk = (const float*)d_in[5];
  const float* bk = (const float*)d_in[6];
  const float* Wv = (const float*)d_in[7];
  const float* bv = (const float*)d_in[8];
  float* out = (float*)d_out;

  const size_t per = (size_t)BATCH * SEQ * DIM;   // 8M bf16 elems each
  bf16_t* qw = (bf16_t*)d_ws;
  bf16_t* kw = qw + per;
  bf16_t* vw = kw + per;

  dim3 g1((BATCH * SEQ) / BM, DIM / BN);          // 64 x 8
  qkv_gemm_kernel<0, 1><<<g1, 128, 0, stream>>>(query, Wq, bq, qw);
  qkv_gemm_kernel<0, 0><<<g1, 128, 0, stream>>>(key_,  Wk, bk, kw);
  qkv_gemm_kernel<1, 0><<<g1, 128, 0, stream>>>(value, Wv, bv, vw);

  dim3 g2(SEQ / 64, BATCH * HEADS);               // 32 x 64
  flash_attn_kernel<<<g2, 128, 0, stream>>>(qw, kw, vw, out);
}
